// DDRSA_7232724926797
// MI455X (gfx1250) — matrix operand order repaired and checked by reading, not compiled, not run on hardware
//
#include <hip/hip_runtime.h>
#include <hip/hip_bf16.h>

// ---------------------------------------------------------------------------
// DDRSA forward for MI455X (gfx1250, wave32, WMMA f16 16x16x32, f32 accum)
//  - encoder gate weights staged in LDS (192KB matrix fits in 320KB WGP LDS)
//  - all LDS rows padded so the 16 row-lanes of each fragment load hit
//    distinct banks (row stride in dwords coprime-ish with 64)
//  - x pre-converted to f16; per-step tile staged with b128 copies + prefetch
// ---------------------------------------------------------------------------
#define B_     512
#define T_     256
#define F_     64
#define H_     128
#define D_     192      // F + H
#define G4H    512      // 4*H
#define G4D    768      // 4*D
#define OUTD   32
#define STEPS  64       // OUT_DIM * RISKS
#define RISKS_ 2
#define NEGV   (-1e10f)
#define BT     32       // batch tile rows per workgroup

// padded LDS row lengths (halves); all chosen so (row*ld_dwords mod 64) is a
// permutation over 16 consecutive rows and row stride stays 16B-aligned.
#define XLD    72       // x tile rows   (36 dwords)
#define HLD    136      // h state rows  (68 dwords)
#define WLD    200      // weight rows   (100 dwords)
#define DLD    392      // decoder [ha|h] rows (196 dwords)

typedef __attribute__((ext_vector_type(16))) _Float16 v16h;
typedef __attribute__((ext_vector_type(8)))  _Float16 v8h;
typedef __attribute__((ext_vector_type(8)))  float    v8f;

__device__ inline v16h make_frag(v8h lo, v8h hi) {
  union { v16h v; v8h h[2]; } u;
  u.h[0] = lo; u.h[1] = hi;
  return u.v;
}

__device__ inline v8f vzero8() {
  v8f z;
#pragma unroll
  for (int i = 0; i < 8; ++i) z[i] = 0.0f;
  return z;
}

// A fragment (16x32 f16): act row-major [16][ld] halves starting at base.
// lane l: m = l&15; halves 0..7 -> K = h8..h8+7 ; halves 8..15 -> K = 16+h8..
__device__ inline v16h load_a_frag(const _Float16* base, int ld, int lane, int kOff) {
  const int m  = lane & 15;
  const int h8 = (lane >> 4) << 3;
  const _Float16* p = base + m * ld + kOff + h8;
  v8h lo = *(const v8h*)(p);
  v8h hi = *(const v8h*)(p + 16);
  return make_frag(lo, hi);
}

// B fragment (32x16 f16): weights row-major [N][ldk] (PyTorch layout; the
// B-matrix of gate = act @ W^T wants contiguous-k per output row n).
__device__ inline v16h load_b_frag(const _Float16* w, int ldk, int lane, int nBase, int kOff) {
  const int n = nBase + (lane & 15);
  const _Float16* p = w + (size_t)n * ldk + kOff + ((lane >> 4) << 3);
  v8h lo = *(const v8h*)(p);
  v8h hi = *(const v8h*)(p + 16);
  return make_frag(lo, hi);
}

__device__ inline v8f wmma_f16(v16h a, v16h b, v8f c) {
  return __builtin_amdgcn_wmma_f32_16x16x32_f16(false, a, false, b, (short)0, c,
                                                false, false);
}

__device__ inline float sigmoidf_(float x) { return 1.0f / (1.0f + __expf(-x)); }
__device__ inline float tanhf_(float x) {
  float e = __expf(2.0f * x);
  return (e - 1.0f) / (e + 1.0f);
}

// ---------------------------------------------------------------------------
// Prep: f32->f16 for x and all matmul weights; concatenate [Wih|Whh] along K.
// ---------------------------------------------------------------------------
__global__ __launch_bounds__(256)
void prep_kernel(const float* __restrict__ x,
                 const float* __restrict__ WihE, const float* __restrict__ WhhE,
                 const float* __restrict__ Wlong,
                 const float* __restrict__ WihCs, const float* __restrict__ WhhCs,
                 const float* __restrict__ Wcs,
                 _Float16* __restrict__ xh,      // [B,T,F]
                 _Float16* __restrict__ WcatE,   // [4H][D]
                 _Float16* __restrict__ WlongH,  // [F][H]
                 _Float16* __restrict__ WcatCs,  // [2][4D][2D]
                 _Float16* __restrict__ WcsH)    // [2][D]
{
  long idx = (long)blockIdx.x * blockDim.x + threadIdx.x;
  const long N0 = (long)B_ * T_ * F_;
  if (idx < N0) { xh[idx] = (_Float16)x[idx]; return; }
  idx -= N0;
  const long N1 = G4H * D_;
  if (idx < N1) {
    int n = (int)(idx / D_), k = (int)(idx % D_);
    float v = (k < F_) ? WihE[n * F_ + k] : WhhE[n * H_ + (k - F_)];
    WcatE[idx] = (_Float16)v;
    return;
  }
  idx -= N1;
  const long N2 = F_ * H_;
  if (idx < N2) { WlongH[idx] = (_Float16)Wlong[idx]; return; }
  idx -= N2;
  const long N3 = (long)RISKS_ * G4D * (2 * D_);
  if (idx < N3) {
    int r = (int)(idx / (G4D * 2 * D_));
    int rem = (int)(idx % (G4D * 2 * D_));
    int n = rem / (2 * D_), k = rem % (2 * D_);
    float v = (k < D_) ? WihCs[((size_t)r * G4D + n) * D_ + k]
                       : WhhCs[((size_t)r * G4D + n) * D_ + (k - D_)];
    WcatCs[idx] = (_Float16)v;
    return;
  }
  idx -= N3;
  if (idx < RISKS_ * D_) WcsH[idx] = (_Float16)Wcs[idx];
}

// ---------------------------------------------------------------------------
// Encoder: 16 persistent WGs x 32 batch rows, t = 0..255.  Gate weights live
// in LDS (one-time 192KB stage); per-step critical path is ds_load -> wmma.
// 8 waves; wave w owns hidden units [16w,16w+16) -> its 4 gate N-tiles.
// ---------------------------------------------------------------------------
__global__ __launch_bounds__(256)
void enc_kernel(const _Float16* __restrict__ xh,      // [B,T,F] f16
                const _Float16* __restrict__ Wcat,    // [4H][D]
                const _Float16* __restrict__ Wlg,     // [F][H]
                const float* __restrict__ blong,      // [F]
                _Float16* __restrict__ hiddenG,       // [B,T,H] f16
                float* __restrict__ outLong)          // d_out [B,T,F]
{
  __shared__ __align__(16) _Float16 wlds[G4H][WLD];   // 204,800 B
  __shared__ __align__(16) _Float16 xbuf[BT][XLD];    //   4,608 B
  __shared__ __align__(16) _Float16 hbuf[BT][HLD];    //   8,704 B

  const int b0   = blockIdx.x * BT;
  const int tid  = threadIdx.x;
  const int wave = tid >> 5;   // 0..7
  const int lane = tid & 31;
  const int n    = wave * 16 + (lane & 15);

  // one-time weight stage into LDS (b128 chunks, padded rows)
  for (int i = tid; i < G4H * (D_ / 8); i += blockDim.x) {
    int r = i / (D_ / 8), c = i % (D_ / 8);
    *(v8h*)(&wlds[r][c * 8]) = *(const v8h*)(Wcat + (size_t)r * D_ + c * 8);
  }
  // h_0 = 0
  for (int i = tid; i < BT * H_; i += blockDim.x)
    hbuf[i / H_][i % H_] = (_Float16)0.0f;

  v8f cst[2];
  cst[0] = vzero8(); cst[1] = vzero8();

  for (int t = 0; t < T_; ++t) {
    __syncthreads();
    // stage x_t tile (256 b128 copies) + prefetch next tile
    {
      int m = tid >> 3, c = tid & 7;
      const _Float16* src = xh + ((size_t)(b0 + m) * T_ + t) * F_ + c * 8;
      *(v8h*)(&xbuf[m][c * 8]) = *(const v8h*)(src);
      if (t + 1 < T_)
        __builtin_prefetch(src + F_, 0, 0);   // next timestep, same row
    }
    __syncthreads();

    // gates = [x_t|h] @ [Wih|Whh]^T   (M=32, N=512, K=192)
    v8f acc[2][4];
#pragma unroll
    for (int mt = 0; mt < 2; ++mt)
#pragma unroll
      for (int g = 0; g < 4; ++g) acc[mt][g] = vzero8();

#pragma unroll
    for (int kt = 0; kt < 6; ++kt) {
      const _Float16* a_base0 = (kt < 2) ? &xbuf[0][0]  : &hbuf[0][0];
      const _Float16* a_base1 = (kt < 2) ? &xbuf[16][0] : &hbuf[16][0];
      const int a_ld  = (kt < 2) ? XLD : HLD;
      const int a_off = (kt < 2) ? kt * 32 : kt * 32 - F_;
      v16h a0 = load_a_frag(a_base0, a_ld, lane, a_off);
      v16h a1 = load_a_frag(a_base1, a_ld, lane, a_off);
#pragma unroll
      for (int g = 0; g < 4; ++g) {
        v16h bf = load_b_frag(&wlds[0][0], WLD, lane, g * H_ + wave * 16, kt * 32);
        acc[0][g] = wmma_f16(a0, bf, acc[0][g]);
        acc[1][g] = wmma_f16(a1, bf, acc[1][g]);
      }
    }
    __syncthreads();   // all reads of hbuf done before h overwrite

    // LSTM nonlinearity (gate order i,f,g,o); c-state stays in registers
#pragma unroll
    for (int mt = 0; mt < 2; ++mt) {
#pragma unroll
      for (int j = 0; j < 8; ++j) {
        int m = mt * 16 + ((lane >> 4) << 3) + j;
        float ig = sigmoidf_(acc[mt][0][j]);
        float fg = sigmoidf_(acc[mt][1][j]);
        float gg = tanhf_(acc[mt][2][j]);
        float og = sigmoidf_(acc[mt][3][j]);
        float cv = fg * cst[mt][j] + ig * gg;
        cst[mt][j] = cv;
        hbuf[m][n] = (_Float16)(og * tanhf_(cv));
      }
    }
    __syncthreads();

    // coalesced hidden dump (16B chunks)
    {
      int m = tid >> 4, c = tid & 15;   // 32 rows x 16 chunks = 512 == 2*256
      v8h v0 = *(const v8h*)(&hbuf[m][c * 8]);
      *(v8h*)(hiddenG + ((size_t)(b0 + m) * T_ + t) * H_ + c * 8) = v0;
      m += 16;
      v8h v1 = *(const v8h*)(&hbuf[m][c * 8]);
      *(v8h*)(hiddenG + ((size_t)(b0 + m) * T_ + t) * H_ + c * 8) = v1;
    }

    // long_pred = h_t @ W_long^T + b  (M=32, N=64, K=128); 8 tiles / 8 waves
    {
      int mt = wave >> 2, nt = wave & 3;
      v8f lacc = vzero8();
#pragma unroll
      for (int kt = 0; kt < 4; ++kt) {
        v16h a  = load_a_frag(&hbuf[mt * 16][0], HLD, lane, kt * 32);
        v16h bf = load_b_frag(Wlg, H_, lane, nt * 16, kt * 32);  // hoists to regs
        lacc = wmma_f16(a, bf, lacc);
      }
      int nn = nt * 16 + (lane & 15);
      float bias = blong[nn];
#pragma unroll
      for (int j = 0; j < 8; ++j) {
        int m = mt * 16 + ((lane >> 4) << 3) + j;
        outLong[((size_t)(b0 + m) * T_ + t) * F_ + nn] = lacc[j] + bias;
      }
    }
  }
}

// ---------------------------------------------------------------------------
// Attention: one WG per batch row (mask, last_obs, masked softmax, weighted
// sum); writes ha = [hidden_att | x_last] f16.
// ---------------------------------------------------------------------------
__global__ __launch_bounds__(256)
void att_kernel(const float* __restrict__ x,
                const _Float16* __restrict__ hiddenG,
                const float* __restrict__ Watt,     // [wah | wax]
                const float* __restrict__ batt,
                _Float16* __restrict__ haG)         // [B][192]
{
  __shared__ float sc[T_];
  __shared__ float red[256];
  __shared__ float xl[F_];
  __shared__ int lastSh;

  const int b = blockIdx.x, tid = threadIdx.x;

  const float* xrow = x + ((size_t)b * T_ + tid) * F_;
  float asum = 0.0f;
  for (int f = 0; f < F_; ++f) asum += fabsf(xrow[f]);
  red[tid] = (asum != 0.0f) ? 1.0f : 0.0f;
  __syncthreads();
  for (int off = 128; off > 0; off >>= 1) {
    if (tid < off) red[tid] += red[tid + off];
    __syncthreads();
  }
  if (tid == 0) lastSh = (int)red[0] - 1;
  __syncthreads();
  const int last = lastSh;
  const int li = ((last % T_) + T_) % T_;   // JAX-style negative wrap

  if (tid < F_) xl[tid] = x[((size_t)b * T_ + li) * F_ + tid];
  __syncthreads();

  float xdot = batt[0];
  for (int f = 0; f < F_; ++f) xdot += xl[f] * Watt[H_ + f];

  const _Float16* hrow = hiddenG + ((size_t)b * T_ + tid) * H_;
  float sco = xdot;
  for (int h = 0; h < H_; ++h) sco += (float)hrow[h] * Watt[h];
  if (tid >= last) sco = NEGV;

  red[tid] = sco;
  __syncthreads();
  for (int off = 128; off > 0; off >>= 1) {
    if (tid < off) red[tid] = fmaxf(red[tid], red[tid + off]);
    __syncthreads();
  }
  const float mx = red[0];
  __syncthreads();
  const float e = __expf(sco - mx);
  red[tid] = e;
  __syncthreads();
  for (int off = 128; off > 0; off >>= 1) {
    if (tid < off) red[tid] += red[tid + off];
    __syncthreads();
  }
  const float attv = (last > 0) ? (e / red[0]) : 0.0f;
  __syncthreads();
  sc[tid] = attv;
  __syncthreads();

  if (tid < H_) {
    float acc = 0.0f;
    for (int t = 0; t < T_; ++t)
      acc += sc[t] * (float)hiddenG[((size_t)b * T_ + t) * H_ + tid];
    haG[(size_t)b * D_ + tid] = (_Float16)acc;
  }
  if (tid < F_) haG[(size_t)b * D_ + H_ + tid] = (_Float16)xl[tid];
}

// ---------------------------------------------------------------------------
// Decoder: 16 WGs x 32 rows, 64 interleaved steps (r = s&1), shared carry,
// input always ha, h0 = c0 = ha.  12 waves; wave w owns D-units [16w,16w+16).
// Gate weights stream from L2 (1.15MB resident) with prefetch hints.
// ---------------------------------------------------------------------------
__global__ __launch_bounds__(384)
void dec_kernel(const _Float16* __restrict__ haG,     // [B][192]
                const _Float16* __restrict__ WcatCs,  // [2][768][384]
                const _Float16* __restrict__ WcsG,    // [2][192]
                const float* __restrict__ bcs,        // [2][1]
                float* __restrict__ logitsG)          // [B][64] sigmoid outputs
{
  __shared__ __align__(16) _Float16 act[BT][DLD];     // [ha | h], padded rows
  __shared__ float lred[BT][8];
  const int b0   = blockIdx.x * BT;
  const int tid  = threadIdx.x;
  const int wave = tid >> 5;   // 0..11
  const int lane = tid & 31;
  const int n    = wave * 16 + (lane & 15);

  for (int i = tid; i < BT * D_; i += blockDim.x) {
    int m = i / D_, k = i % D_;
    _Float16 v = haG[(size_t)(b0 + m) * D_ + k];
    act[m][k]      = v;   // constant input half
    act[m][D_ + k] = v;   // h0 = ha
  }
  __syncthreads();

  // c0 = ha in fragment layout
  v8f cst[2];
#pragma unroll
  for (int mt = 0; mt < 2; ++mt)
#pragma unroll
    for (int j = 0; j < 8; ++j) {
      int m = mt * 16 + ((lane >> 4) << 3) + j;
      cst[mt][j] = (float)act[m][n];
    }

  for (int s = 0; s < STEPS; ++s) {
    const int r = s & 1;
    const _Float16* W = WcatCs + (size_t)r * G4D * (2 * D_);
    // warm L2/L0 for the other risk's slice this wave will need next step
    {
      const _Float16* Wn = WcatCs + (size_t)(1 - r) * G4D * (2 * D_);
      __builtin_prefetch(Wn + (size_t)(wave * 16 + (lane & 15)) * (2 * D_), 0, 1);
    }

    v8f acc[2][4];
#pragma unroll
    for (int mt = 0; mt < 2; ++mt)
#pragma unroll
      for (int g = 0; g < 4; ++g) acc[mt][g] = vzero8();

    for (int kt = 0; kt < 12; ++kt) {
      v16h a0 = load_a_frag(&act[0][0],  DLD, lane, kt * 32);
      v16h a1 = load_a_frag(&act[16][0], DLD, lane, kt * 32);
#pragma unroll
      for (int g = 0; g < 4; ++g) {
        v16h bf = load_b_frag(W, 2 * D_, lane, g * D_ + wave * 16, kt * 32);
        acc[0][g] = wmma_f16(a0, bf, acc[0][g]);
        acc[1][g] = wmma_f16(a1, bf, acc[1][g]);
      }
    }
    __syncthreads();

#pragma unroll
    for (int mt = 0; mt < 2; ++mt) {
#pragma unroll
      for (int j = 0; j < 8; ++j) {
        int m = mt * 16 + ((lane >> 4) << 3) + j;
        float ig = sigmoidf_(acc[mt][0][j]);
        float fg = sigmoidf_(acc[mt][1][j]);
        float gg = tanhf_(acc[mt][2][j]);
        float og = sigmoidf_(acc[mt][3][j]);
        float cv = fg * cst[mt][j] + ig * gg;
        cst[mt][j] = cv;
        act[m][D_ + n] = (_Float16)(og * tanhf_(cv));
      }
    }
    __syncthreads();

    // out_s[b] = sigmoid(h . W_cs[r] + b_cs[r])  (parallel partial dots)
    if (tid < BT * 8) {
      int m = tid >> 3, sgi = tid & 7;
      const _Float16* Wc = WcsG + r * D_;
      float p = 0.0f;
      for (int k = sgi * 24; k < sgi * 24 + 24; ++k)
        p += (float)act[m][D_ + k] * (float)Wc[k];
      lred[m][sgi] = p;
    }
    __syncthreads();
    if (tid < BT) {
      float a2 = bcs[r];
#pragma unroll
      for (int q = 0; q < 8; ++q) a2 += lred[tid][q];
      logitsG[(size_t)(b0 + tid) * STEPS + s] = sigmoidf_(a2);
    }
    __syncthreads();
  }
}

// ---------------------------------------------------------------------------
// Final: softmax over the 64 sigmoid outputs; scatter to the two risk parts.
// ---------------------------------------------------------------------------
__global__ __launch_bounds__(64)
void fin_kernel(const float* __restrict__ logitsG, float* __restrict__ out)
{
  __shared__ float red[64];
  const int b = blockIdx.x, tid = threadIdx.x;   // tid == step s
  const float v = logitsG[(size_t)b * STEPS + tid];
  red[tid] = v;
  __syncthreads();
  for (int off = 32; off > 0; off >>= 1) {
    if (tid < off) red[tid] = fmaxf(red[tid], red[tid + off]);
    __syncthreads();
  }
  const float mx = red[0];
  __syncthreads();
  const float e = __expf(v - mx);
  red[tid] = e;
  __syncthreads();
  for (int off = 32; off > 0; off >>= 1) {
    if (tid < off) red[tid] += red[tid + off];
    __syncthreads();
  }
  const float sm = e / red[0];
  const size_t base = (size_t)B_ * T_ * F_;
  out[base + (size_t)(tid >> 5) * (B_ * OUTD) + (size_t)b * OUTD + (tid & 31)] = sm;
}

// ---------------------------------------------------------------------------
extern "C" void kernel_launch(void* const* d_in, const int* in_sizes, int n_in,
                              void* d_out, int out_size, void* d_ws, size_t ws_size,
                              hipStream_t stream)
{
  const float* x     = (const float*)d_in[0];
  const float* WihE  = (const float*)d_in[1];
  const float* WhhE  = (const float*)d_in[2];
  const float* Wlong = (const float*)d_in[3];
  const float* blong = (const float*)d_in[4];
  const float* Watt  = (const float*)d_in[5];
  const float* batt  = (const float*)d_in[6];
  const float* WihCs = (const float*)d_in[7];
  const float* WhhCs = (const float*)d_in[8];
  const float* Wcs   = (const float*)d_in[9];
  const float* bcs   = (const float*)d_in[10];

  char* ws = (char*)d_ws;
  size_t off = 0;
  _Float16* hiddenG = (_Float16*)(ws + off); off += (size_t)B_ * T_ * H_ * sizeof(_Float16);
  _Float16* xhG     = (_Float16*)(ws + off); off += (size_t)B_ * T_ * F_ * sizeof(_Float16);
  _Float16* haG     = (_Float16*)(ws + off); off += (size_t)B_ * D_ * sizeof(_Float16);
  _Float16* WcatE   = (_Float16*)(ws + off); off += (size_t)G4H * D_ * sizeof(_Float16);
  _Float16* WlongH  = (_Float16*)(ws + off); off += (size_t)F_ * H_ * sizeof(_Float16);
  _Float16* WcatCs  = (_Float16*)(ws + off); off += (size_t)RISKS_ * G4D * 2 * D_ * sizeof(_Float16);
  _Float16* WcsH    = (_Float16*)(ws + off); off += (size_t)RISKS_ * D_ * sizeof(_Float16);
  float*    logitsG = (float*)(ws + off);    off += (size_t)B_ * STEPS * sizeof(float);

  const long prepTotal = (long)B_ * T_ * F_ + G4H * D_ + F_ * H_ +
                         (long)RISKS_ * G4D * 2 * D_ + RISKS_ * D_;
  prep_kernel<<<(int)((prepTotal + 255) / 256), 256, 0, stream>>>(
      x, WihE, WhhE, Wlong, WihCs, WhhCs, Wcs, xhG, WcatE, WlongH, WcatCs, WcsH);

  enc_kernel<<<B_ / BT, 256, 0, stream>>>(xhG, WcatE, WlongH, blong, hiddenG,
                                          (float*)d_out);
  att_kernel<<<B_, 256, 0, stream>>>(x, hiddenG, Watt, batt, haG);
  dec_kernel<<<B_ / BT, 384, 0, stream>>>(haG, WcatCs, WcsH, bcs, logitsG);
  fin_kernel<<<B_, 64, 0, stream>>>(logitsG, (float*)d_out);

  (void)in_sizes; (void)n_in; (void)out_size; (void)ws_size;
}
